// minerva_35124242547419
// MI455X (gfx1250) — compile-verified, matrix operand order checked by reading, not run on hardware
//
#include <hip/hip_runtime.h>
#include <math.h>

typedef __attribute__((ext_vector_type(16))) _Float16 v16h;
typedef __attribute__((ext_vector_type(8)))  _Float16 v8h;
typedef __attribute__((ext_vector_type(8)))  float    v8f;

#define BQ     4096
#define NEX    16384
#define DIN    1024
#define DF     256
#define NLAB   28
#define DCLS   64
#define MTOT   (BQ + NEX)        // 20480 rows projected (queries then exemplars)
#define SPLIT  8
#define EX_PER_SPLIT (NEX / SPLIT)  // 2048

__device__ __forceinline__ v16h mk16(v8h a, v8h b) {
  return __builtin_shufflevector(a, b, 0,1,2,3,4,5,6,7,8,9,10,11,12,13,14,15);
}

// ---------------------------------------------------------------- converts
__global__ void cvt_f32_f16(const float* __restrict__ src,
                            _Float16* __restrict__ dst, int n) {
  int i = blockIdx.x * blockDim.x + threadIdx.x;
  const int stride = gridDim.x * blockDim.x;
  for (; i < n; i += stride) dst[i] = (_Float16)src[i];
}

// ---------------------------------------------------------------- projection GEMM
// P[MTOT, 256] = X16[MTOT, 1024] @ GW16^T   (B column n = g_w row n, contiguous)
// one wave per 16x16 output tile, K=1024 -> 32 WMMAs
__global__ void __launch_bounds__(32) proj_wmma(const _Float16* __restrict__ x16,
                                                const _Float16* __restrict__ gw16,
                                                float* __restrict__ proj) {
  const int lane  = threadIdx.x & 31;
  const int lrow  = lane & 15;
  const int khalf = (lane >> 4) << 3;   // 0 or 8
  const int tm = blockIdx.x, tn = blockIdx.y;
  const _Float16* Ar = x16  + (size_t)(tm * 16 + lrow) * DIN;
  const _Float16* Br = gw16 + (size_t)(tn * 16 + lrow) * DIN;
  v8f acc = {};
#pragma unroll 4
  for (int k0 = 0; k0 < DIN; k0 += 32) {
    v16h a = mk16(*(const v8h*)(Ar + k0 + khalf), *(const v8h*)(Ar + k0 + khalf + 16));
    v16h b = mk16(*(const v8h*)(Br + k0 + khalf), *(const v8h*)(Br + k0 + khalf + 16));
    acc = __builtin_amdgcn_wmma_f32_16x16x32_f16(false, a, false, b, (short)0, acc,
                                                 false, false);
  }
#pragma unroll
  for (int k = 0; k < 8; ++k)
    proj[(size_t)(tm * 16 + khalf + k) * DF + tn * 16 + lrow] = acc[k];
}

// ---------------------------------------------------------------- per-row L2 norm
// one wave per row of 256 fp32; write normalized f16 row
__global__ void __launch_bounds__(32) l2norm_rows(const float* __restrict__ proj,
                                                  _Float16* __restrict__ xn16) {
  const int row  = blockIdx.x;
  const int lane = threadIdx.x & 31;
  const float* r = proj + (size_t)row * DF;
  float v[8];
  float ss = 0.f;
#pragma unroll
  for (int i = 0; i < 8; ++i) { v[i] = r[lane * 8 + i]; ss += v[i] * v[i]; }
#pragma unroll
  for (int m = 16; m >= 1; m >>= 1) ss += __shfl_xor(ss, m, 32);
  const float inv = 1.f / fmaxf(sqrtf(ss), 1e-12f);
  _Float16* o = xn16 + (size_t)row * DF;
#pragma unroll
  for (int i = 0; i < 8; ++i) o[lane * 8 + i] = (_Float16)(v[i] * inv);
}

// ---------------------------------------------------------------- exemplar class reps
// ecrT[d][e] = (L1norm(ex_classes[e]) @ class_reps)[d]   stored transposed, f16
__global__ void __launch_bounds__(64) ecr_kernel(const float* __restrict__ ex_classes,
                                                 const float* __restrict__ class_reps,
                                                 _Float16* __restrict__ ecrT) {
  __shared__ float cr[NLAB * DCLS];
  const int t = threadIdx.x;  // 0..63
  for (int i = t; i < NLAB * DCLS; i += 64) cr[i] = class_reps[i];
  __syncthreads();
  const int e = blockIdx.x * 64 + t;
  float u[NLAB];
  float s = 0.f;
#pragma unroll
  for (int c = 0; c < NLAB; ++c) {
    u[c] = ex_classes[(size_t)e * NLAB + c];
    s += fabsf(u[c]);
  }
  const float inv = 1.f / fmaxf(s, 1e-12f);
  for (int d = 0; d < DCLS; ++d) {
    float acc = 0.f;
#pragma unroll
    for (int c = 0; c < NLAB; ++c) acc = fmaf(u[c], cr[c * DCLS + d], acc);
    ecrT[(size_t)d * NEX + e] = (_Float16)(acc * inv);
  }
}

// ---------------------------------------------------------------- fused sim^3 + echo
// WG = 4 waves, 64 query rows, one exemplar split of 2048.
// per 64-exemplar chunk: 32 sim WMMAs + cube + LDS relayout + 8 echo WMMAs per wave.
__global__ void __launch_bounds__(128) fused_sim_echo(
    const _Float16* __restrict__ xn16,   // [MTOT, 256] normalized f16
    const _Float16* __restrict__ ecrT,   // [DCLS, NEX] f16
    float* __restrict__ rowsum_part,     // [SPLIT, BQ]
    float* __restrict__ echo_part) {     // [SPLIT, BQ, DCLS]
  __shared__ __align__(16) _Float16 wlds[4][16][64];
  const int lane  = threadIdx.x & 31;
  const int wid   = threadIdx.x >> 5;
  const int lrow  = lane & 15;
  const int khalf = (lane >> 4) << 3;    // 0 or 8
  const int m0    = blockIdx.x * 64;
  const int split = blockIdx.y;
  const int exBeg = split * EX_PER_SPLIT;
  const int exEnd = exBeg + EX_PER_SPLIT;

  // query A-fragments, K = 256 -> 8 fragments held for the whole loop
  const _Float16* qr = xn16 + (size_t)(m0 + wid * 16 + lrow) * DF;
  v16h qa[8];
#pragma unroll
  for (int ks = 0; ks < 8; ++ks) {
    const int kb = ks * 32 + khalf;
    qa[ks] = mk16(*(const v8h*)(qr + kb), *(const v8h*)(qr + kb + 16));
  }

  v8f eacc[4] = {{}, {}, {}, {}};
  v8f rs = {};

  for (int e0 = exBeg; e0 < exEnd; e0 += 64) {
    if (e0 + 64 < exEnd) {  // prefetch next exemplar chunk (global_prefetch_b8)
      const _Float16* pf =
          xn16 + (size_t)(BQ + e0 + 64 + (threadIdx.x >> 1)) * DF + (threadIdx.x & 1) * 128;
      __builtin_prefetch(pf, 0, 3);
    }
    // ---- similarity tiles: S[16 x 64], K = 256
#pragma unroll
    for (int ct = 0; ct < 4; ++ct) {
      const _Float16* br = xn16 + (size_t)(BQ + e0 + ct * 16 + lrow) * DF;
      v8f sacc = {};
#pragma unroll
      for (int ks = 0; ks < 8; ++ks) {
        const int kb = ks * 32 + khalf;
        v16h b = mk16(*(const v8h*)(br + kb), *(const v8h*)(br + kb + 16));
        sacc = __builtin_amdgcn_wmma_f32_16x16x32_f16(false, qa[ks], false, b, (short)0,
                                                      sacc, false, false);
      }
      // w = sign(s)|s|^3 = s^3 ; |w| accumulates into L1 rowsum partials
#pragma unroll
      for (int k = 0; k < 8; ++k) {
        const float s = sacc[k];
        const float w = s * s * s;
        rs[k] += fabsf(w);
        wlds[wid][khalf + k][ct * 16 + lrow] = (_Float16)w;
      }
    }
    __syncthreads();
    // ---- echo tiles: E[16 x 64] += W[16 x 64] @ ECR[64 x 64]
#pragma unroll
    for (int s2 = 0; s2 < 2; ++s2) {
      const int kb = s2 * 32 + khalf;
      v16h wa = mk16(*(const v8h*)&wlds[wid][lrow][kb],
                     *(const v8h*)&wlds[wid][lrow][kb + 16]);
#pragma unroll
      for (int ct = 0; ct < 4; ++ct) {
        const _Float16* er = ecrT + (size_t)(ct * 16 + lrow) * NEX + e0;
        v16h b = mk16(*(const v8h*)(er + kb), *(const v8h*)(er + kb + 16));
        eacc[ct] = __builtin_amdgcn_wmma_f32_16x16x32_f16(false, wa, false, b, (short)0,
                                                          eacc[ct], false, false);
      }
    }
    __syncthreads();
  }

  // rowsum: reduce |w| partials across the 16 column-lanes of each half
#pragma unroll
  for (int k = 0; k < 8; ++k) {
    float v = rs[k];
    v += __shfl_xor(v, 1, 32);
    v += __shfl_xor(v, 2, 32);
    v += __shfl_xor(v, 4, 32);
    v += __shfl_xor(v, 8, 32);
    if (lrow == 0)
      rowsum_part[(size_t)split * BQ + m0 + wid * 16 + khalf + k] = v;
  }
  // echo partials: unique writer per (split,row,d) slot -> deterministic
#pragma unroll
  for (int ct = 0; ct < 4; ++ct)
#pragma unroll
    for (int k = 0; k < 8; ++k)
      echo_part[((size_t)split * BQ + m0 + wid * 16 + khalf + k) * DCLS + ct * 16 + lrow] =
          eacc[ct][k];
}

// ---------------------------------------------------------------- finalize per row
__global__ void __launch_bounds__(32) finalize_kernel(
    const float* __restrict__ rowsum_part, const float* __restrict__ echo_part,
    const float* __restrict__ class_reps, const float* __restrict__ labels,
    float* __restrict__ out, float* __restrict__ lossp) {
  const int b    = blockIdx.x;
  const int lane = threadIdx.x & 31;
  float rsum = 0.f;
#pragma unroll
  for (int sp = 0; sp < SPLIT; ++sp) rsum += rowsum_part[(size_t)sp * BQ + b];
  float e0 = 0.f, e1 = 0.f;
#pragma unroll
  for (int sp = 0; sp < SPLIT; ++sp) {
    const float* ep = echo_part + ((size_t)sp * BQ + b) * DCLS;
    e0 += ep[lane];
    e1 += ep[lane + 32];
  }
  const float inv = 1.f / fmaxf(rsum, 1e-12f);
  e0 *= inv;
  e1 *= inv;
  float loss = 0.f;
  for (int l = 0; l < NLAB; ++l) {
    const float d0 = e0 - class_reps[l * DCLS + lane];
    const float d1 = e1 - class_reps[l * DCLS + lane + 32];
    float p = d0 * d0 + d1 * d1;
#pragma unroll
    for (int m = 16; m >= 1; m >>= 1) p += __shfl_xor(p, m, 32);
    const float nd = -sqrtf(p);
    if (lane == 0) out[1 + b * NLAB + l] = nd;
    const float y   = labels[b * NLAB + l];
    const float lse = log1pf(expf(-fabsf(nd)));  // softplus(-|nd|)
    const float lsp = fminf(nd, 0.f) - lse;      // log_sigmoid(nd)
    const float lsn = fminf(-nd, 0.f) - lse;     // log_sigmoid(-nd)
    loss -= y * lsp + (1.f - y) * lsn;
  }
  if (lane == 0) lossp[b] = loss;
}

// fixed-order deterministic loss mean
__global__ void __launch_bounds__(256) loss_reduce(const float* __restrict__ lossp,
                                                   float* __restrict__ out) {
  __shared__ float sm[256];
  const int t = threadIdx.x;
  float s = 0.f;
  for (int i = t; i < BQ; i += 256) s += lossp[i];
  sm[t] = s;
  __syncthreads();
  for (int w = 128; w > 0; w >>= 1) {
    if (t < w) sm[t] += sm[t + w];
    __syncthreads();
  }
  if (t == 0) out[0] = sm[0] / (float)(BQ * NLAB);
}

// ---------------------------------------------------------------- launch
extern "C" void kernel_launch(void* const* d_in, const int* in_sizes, int n_in,
                              void* d_out, int out_size, void* d_ws, size_t ws_size,
                              hipStream_t stream) {
  const float* features    = (const float*)d_in[0];
  const float* labels      = (const float*)d_in[1];
  const float* ex_features = (const float*)d_in[2];
  const float* ex_classes  = (const float*)d_in[3];
  const float* g_w         = (const float*)d_in[4];
  const float* class_reps  = (const float*)d_in[5];
  float* out = (float*)d_out;

  char* ws = (char*)d_ws;
  size_t off = 0;
  auto alloc = [&](size_t bytes) -> void* {
    void* p = ws + off;
    off = (off + bytes + 255) & ~(size_t)255;
    return p;
  };
  _Float16* x16   = (_Float16*)alloc((size_t)MTOT * DIN * 2);
  _Float16* gw16  = (_Float16*)alloc((size_t)DF * DIN * 2);
  float*    proj  = (float*)alloc((size_t)MTOT * DF * 4);
  _Float16* xn16  = (_Float16*)alloc((size_t)MTOT * DF * 2);
  _Float16* ecrT  = (_Float16*)alloc((size_t)DCLS * NEX * 2);
  float*    rsp   = (float*)alloc((size_t)SPLIT * BQ * 4);
  float*    echop = (float*)alloc((size_t)SPLIT * BQ * DCLS * 4);
  float*    lossp = (float*)alloc((size_t)BQ * 4);

  cvt_f32_f16<<<2048, 256, 0, stream>>>(features, x16, BQ * DIN);
  cvt_f32_f16<<<2048, 256, 0, stream>>>(ex_features, x16 + (size_t)BQ * DIN, NEX * DIN);
  cvt_f32_f16<<<512, 256, 0, stream>>>(g_w, gw16, DF * DIN);

  proj_wmma<<<dim3(MTOT / 16, DF / 16), 32, 0, stream>>>(x16, gw16, proj);
  l2norm_rows<<<MTOT, 32, 0, stream>>>(proj, xn16);
  ecr_kernel<<<NEX / 64, 64, 0, stream>>>(ex_classes, class_reps, ecrT);
  fused_sim_echo<<<dim3(BQ / 64, SPLIT), 128, 0, stream>>>(xn16, ecrT, rsp, echop);
  finalize_kernel<<<BQ, 32, 0, stream>>>(rsp, echop, class_reps, labels, out, lossp);
  loss_reduce<<<1, 256, 0, stream>>>(lossp, out);
}
